// GroupedQueryAttention_1073741824356
// MI455X (gfx1250) — matrix operand order repaired and checked by reading, not compiled, not run on hardware
//
#include <hip/hip_runtime.h>

// ---------------------------------------------------------------------------
// GQA attention w/ RoPE for gfx1250 (MI455X), f16 WMMA everywhere.
// Stages: cvt x/w -> f16 (weights transposed)  ->  qkv = x@w_qkv (f16 out)
//         -> RoPE(q,k) -> vT transpose -> causal flash attention (GQA rep=4)
//         -> y = o@w_out (f32 out).
// Data movement: double-buffered async global->LDS (ASYNCcnt pipelined);
// all fragment reads are contiguous 16B runs (ds_load_b128).  Causal mask
// applied only on the diagonal key block.
// ---------------------------------------------------------------------------

typedef __attribute__((ext_vector_type(16))) _Float16 v16h;
typedef __attribute__((ext_vector_type(8)))  float    v8f;
typedef __attribute__((ext_vector_type(4)))  _Float16 v4h;
typedef __attribute__((__vector_size__(16))) int      v4i;

static constexpr int T_    = 2048;
static constexpr int E_    = 2048;
static constexpr int H_    = 32;
static constexpr int Dh    = 64;
static constexpr int KDIM_ = 1024;            // J * Dh
static constexpr int QKVC  = E_ + 2 * KDIM_;  // 4096

#if __has_builtin(__builtin_amdgcn_global_load_async_to_lds_b128)
#define HAVE_ASYNC_LDS 1
#else
#define HAVE_ASYNC_LDS 0
#endif

typedef __attribute__((address_space(1))) v4i* as1_v4i_ptr;
typedef __attribute__((address_space(3))) v4i* as3_v4i_ptr;

// 16-byte global -> LDS copy; async (ASYNCcnt-tracked) when available.
__device__ __forceinline__ void cp16_g2s(_Float16* l, const _Float16* g) {
#if HAVE_ASYNC_LDS
    __builtin_amdgcn_global_load_async_to_lds_b128(
        (as1_v4i_ptr)g, (as3_v4i_ptr)l, 0, 0);
#else
    *(uint4*)l = *(const uint4*)g;
#endif
}

// Wait until at most CNT async loads remain outstanding (in-order retire).
template <int CNT>
__device__ __forceinline__ void stage_wait() {
#if HAVE_ASYNC_LDS
#if __has_builtin(__builtin_amdgcn_s_wait_asynccnt)
    __builtin_amdgcn_s_wait_asynccnt(CNT);
#else
    asm volatile("s_wait_asynccnt %0" :: "i"(CNT) : "memory");
#endif
#endif
}

// ------------------------- one-time f16 conversions ------------------------
__global__ void cvt_f16(const float* __restrict__ s, _Float16* __restrict__ d,
                        int n4) {
    int i = blockIdx.x * blockDim.x + threadIdx.x;
    if (i >= n4) return;
    float4 v = ((const float4*)s)[i];
    v4h h = {(_Float16)v.x, (_Float16)v.y, (_Float16)v.z, (_Float16)v.w};
    *(v4h*)(d + 4 * (size_t)i) = h;
}

// src[R][C] f32 -> dst[C][R] f16 (tiled 32x32 through LDS)
__global__ __launch_bounds__(256) void transpose_cvt(
    const float* __restrict__ src, _Float16* __restrict__ dst, int R, int C) {
    __shared__ _Float16 tile[32][33];
    const int bx = blockIdx.x * 32;   // col
    const int by = blockIdx.y * 32;   // row
    const int tx = threadIdx.x & 31, ty = threadIdx.x >> 5;
#pragma unroll
    for (int i = 0; i < 32; i += 8)
        tile[ty + i][tx] = (_Float16)src[(size_t)(by + ty + i) * C + bx + tx];
    __syncthreads();
#pragma unroll
    for (int i = 0; i < 32; i += 8)
        dst[(size_t)(bx + ty + i) * R + by + tx] = tile[tx][ty + i];
}

// v slice of qkvh[T][4096] (cols 3072..4095) -> vT[1024][T] f16
__global__ __launch_bounds__(256) void transpose_v(
    const _Float16* __restrict__ qkvh, _Float16* __restrict__ vT) {
    __shared__ _Float16 tile[32][33];
    const int bx = blockIdx.x * 32;   // v col (0..1023)
    const int by = blockIdx.y * 32;   // t
    const int tx = threadIdx.x & 31, ty = threadIdx.x >> 5;
#pragma unroll
    for (int i = 0; i < 32; i += 8)
        tile[ty + i][tx] =
            qkvh[(size_t)(by + ty + i) * QKVC + (E_ + KDIM_) + bx + tx];
    __syncthreads();
#pragma unroll
    for (int i = 0; i < 32; i += 8)
        vT[(size_t)(bx + ty + i) * T_ + by + tx] = tile[tx][ty + i];
}

// ------------------------------- RoPE (f16) --------------------------------
__global__ void rope_f16(_Float16* __restrict__ qk,
                         const float* __restrict__ freq) {
    const int pairsPerRow = (E_ + KDIM_) / 2;   // 1536
    int idx = blockIdx.x * blockDim.x + threadIdx.x;
    if (idx >= T_ * pairsPerRow) return;
    const int t   = idx / pairsPerRow;
    const int col = (idx % pairsPerRow) * 2;    // even col in [0, 3072)
    const int f   = (col & (Dh - 1)) >> 1;
    const float c = freq[t * Dh + f * 2 + 0];
    const float s = freq[t * Dh + f * 2 + 1];
    _Float16* p = qk + (size_t)t * QKVC + col;
    const float x0 = (float)p[0], x1 = (float)p[1];
    p[0] = (_Float16)(x0 * c - x1 * s);
    p[1] = (_Float16)(x1 * c + x0 * s);
}

// ------------------------------ WMMA GEMM ----------------------------------
// C[M,N] = A[M,K] @ Bt[N,K]^T.  f16 in, f16 or f32 out.  Block 128x128,
// K-tile 64; 8 waves (2x4), wave tile 64x32 = 4x2 WMMA tiles, 16 wmma/K-tile.
// Double-buffered async staging: tile i+1 in flight while tile i computes.
#define GTM 128
#define GTN 128
#define GTK 64
#define GPAD (GTK + 8)    // 72 halfs; keeps 16B alignment of 8-half runs

template <int N, int K, bool HALF_OUT>
__global__ __launch_bounds__(256) void wmma_gemm(
    const _Float16* __restrict__ A, const _Float16* __restrict__ Bt,
    void* __restrict__ Cout)
{
    __shared__ _Float16 As[2][GTM][GPAD];
    __shared__ _Float16 Bs[2][GTN][GPAD];

    const int tid  = threadIdx.x;
    const int lane = tid & 31;
    const int wave = tid >> 5;
    const int wm   = wave >> 2;       // 0..1
    const int wn   = wave & 3;        // 0..3
    const int m0   = blockIdx.y * GTM;
    const int n0   = blockIdx.x * GTN;
    const int lh   = lane >> 4;
    const int ll   = lane & 15;

    v8f acc[4][2];
#pragma unroll
    for (int i = 0; i < 4; ++i)
#pragma unroll
        for (int j = 0; j < 2; ++j)
            acc[i][j] = (v8f){0.f, 0.f, 0.f, 0.f, 0.f, 0.f, 0.f, 0.f};

    const int sr = tid >> 1;          // staging row 0..127
    const int sc = (tid & 1) * 32;    // staging col 0 / 32 (halfs)

    auto issue = [&](int k0, int buf) {
        const _Float16* ag = A  + (size_t)(m0 + sr) * K + k0 + sc;
        const _Float16* bg = Bt + (size_t)(n0 + sr) * K + k0 + sc;
#pragma unroll
        for (int i = 0; i < 4; ++i) {
            cp16_g2s(&As[buf][sr][sc + i * 8], ag + i * 8);
            cp16_g2s(&Bs[buf][sr][sc + i * 8], bg + i * 8);
        }
    };

    constexpr int NITER = K / GTK;
    issue(0, 0);

    for (int it = 0; it < NITER; ++it) {
        const int buf = it & 1;
        if (it + 1 < NITER) {
            issue((it + 1) * GTK, buf ^ 1);   // prefetch next tile
            stage_wait<8>();                  // retire tile `it` only
        } else {
            stage_wait<0>();
        }
        __syncthreads();

#pragma unroll
        for (int ks = 0; ks < 2; ++ks) {
            v16h bf[2];
#pragma unroll
            for (int nt = 0; nt < 2; ++nt) {
                const int bn = wn * 32 + nt * 16 + ll;
#pragma unroll
                for (int e = 0; e < 16; ++e)
                    bf[nt][e] = Bs[buf][bn][ks * 32 + lh * 16 + e];
            }
#pragma unroll
            for (int mt = 0; mt < 4; ++mt) {
                v16h a;
                const int row = wm * 64 + mt * 16 + ll;
#pragma unroll
                for (int e = 0; e < 16; ++e)
                    a[e] = As[buf][row][ks * 32 + lh * 16 + e];
#pragma unroll
                for (int nt = 0; nt < 2; ++nt)
                    acc[mt][nt] = __builtin_amdgcn_wmma_f32_16x16x32_f16(
                        false, a, false, bf[nt], (short)0, acc[mt][nt], false, false);
            }
        }
        __syncthreads();
    }

#pragma unroll
    for (int mt = 0; mt < 4; ++mt)
#pragma unroll
        for (int nt = 0; nt < 2; ++nt) {
            const int row = m0 + wm * 64 + mt * 16 + lh * 8;
            const int col = n0 + wn * 32 + nt * 16 + ll;
            if constexpr (HALF_OUT) {
                _Float16* C = (_Float16*)Cout;
#pragma unroll
                for (int r = 0; r < 8; ++r)
                    C[(size_t)(row + r) * N + col] = (_Float16)acc[mt][nt][r];
            } else {
                float* C = (float*)Cout;
#pragma unroll
                for (int r = 0; r < 8; ++r)
                    C[(size_t)(row + r) * N + col] = acc[mt][nt][r];
            }
        }
}

// --------------------------- Flash attention -------------------------------
// Grid (T/64 query blocks, H heads); 128 threads = 4 waves, wave w owns query
// rows [qb*64 + w*16, +16).  K tile from qkvh (f16), V tile from vT[d][t].
// Softmax in base-2: 1/sqrt(D) * log2(e) folded into the resident Q fragment.
// Mask applied only on the diagonal key block (kb == qb).
#define QB 64
#define KB 64
#define KPAD (Dh + 8)
#define PPAD (KB + 8)

__global__ __launch_bounds__(128) void flash_attn_wmma(
    const _Float16* __restrict__ qkvh, const _Float16* __restrict__ vT,
    _Float16* __restrict__ oh)
{
    __shared__ _Float16 Ks[2][KB][KPAD];   // [key][d]
    __shared__ _Float16 Vt[2][Dh][PPAD];   // [d][key]
    __shared__ _Float16 Ps[4][16][PPAD];   // per-wave P tile [row][key]

    const int tid  = threadIdx.x;
    const int lane = tid & 31;
    const int w    = tid >> 5;
    const int lh   = lane >> 4;
    const int ll   = lane & 15;
    const int qb   = blockIdx.x;
    const int h    = blockIdx.y;
    const int j    = h >> 2;              // kv head (rep = 4)
    const int qcol = h * Dh;
    const int kcol = E_ + j * Dh;
    const int qrow0 = qb * QB + w * 16;

    const int sr = tid >> 1;              // 0..63
    const int sc = (tid & 1) * 32;        // 0 / 32

    auto issue = [&](int kb, int buf) {
        const _Float16* kg = qkvh + (size_t)(kb * KB + sr) * QKVC + kcol + sc;
        const _Float16* vg = vT + (size_t)(j * Dh + sr) * T_ + kb * KB + sc;
#pragma unroll
        for (int i = 0; i < 4; ++i) {
            cp16_g2s(&Ks[buf][sr][sc + i * 8], kg + i * 8);
            cp16_g2s(&Vt[buf][sr][sc + i * 8], vg + i * 8);
        }
    };
    issue(0, 0);

    // resident Q fragments (A layout); fold (1/sqrt(64))*log2(e) so the
    // softmax can run in base 2 (hardware v_exp_f32 is base-2).
    const _Float16 qs = (_Float16)0.18033688f;   // 0.125 * 1.4426950
    v16h qf[2];
    {
        const _Float16* qp = qkvh + (size_t)(qrow0 + ll) * QKVC + qcol;
#pragma unroll
        for (int s = 0; s < 2; ++s)
#pragma unroll
            for (int e = 0; e < 16; ++e)
                qf[s][e] = qp[s * 32 + lh * 16 + e] * qs;
    }

    v8f oacc[4];
#pragma unroll
    for (int nt = 0; nt < 4; ++nt)
        oacc[nt] = (v8f){0.f, 0.f, 0.f, 0.f, 0.f, 0.f, 0.f, 0.f};
    float mrow[8], lrow[8];
#pragma unroll
    for (int r = 0; r < 8; ++r) { mrow[r] = -1e30f; lrow[r] = 0.f; }

    for (int kb = 0; kb <= qb; ++kb) {
        const int buf = kb & 1;
        if (kb + 1 <= qb) {
            issue(kb + 1, buf ^ 1);       // prefetch next K/V tile
            stage_wait<8>();              // retire tile `kb` only
        } else {
            stage_wait<0>();
        }
        __syncthreads();

        // ---- S = Q @ K^T : 4 key sub-tiles, contraction over D = 64 ----
        v8f sa[4];
#pragma unroll
        for (int nt = 0; nt < 4; ++nt) {
            v8f s = (v8f){0.f, 0.f, 0.f, 0.f, 0.f, 0.f, 0.f, 0.f};
#pragma unroll
            for (int ks = 0; ks < 2; ++ks) {
                v16h b;   // lane = key (N), element -> d (K); contiguous run
#pragma unroll
                for (int e = 0; e < 16; ++e)
                    b[e] = Ks[buf][nt * 16 + ll][ks * 32 + lh * 16 + e];
                s = __builtin_amdgcn_wmma_f32_16x16x32_f16(
                    false, qf[ks], false, b, (short)0, s, false, false);
            }
            sa[nt] = s;
        }

        // ---- causal mask: only the diagonal block needs it ----
        if (kb == qb) {
            const int qg = qrow0 + lh * 8;
#pragma unroll
            for (int nt = 0; nt < 4; ++nt) {
                const int keyIdx = kb * KB + nt * 16 + ll;
#pragma unroll
                for (int r = 0; r < 8; ++r)
                    sa[nt][r] = (keyIdx > qg + r) ? -1e30f : sa[nt][r];
            }
        }

        // ---- online softmax (base 2) ----
        float alpha[8];
#pragma unroll
        for (int r = 0; r < 8; ++r) {
            float v = fmaxf(fmaxf(sa[0][r], sa[1][r]), fmaxf(sa[2][r], sa[3][r]));
#pragma unroll
            for (int off = 8; off >= 1; off >>= 1)
                v = fmaxf(v, __shfl_xor(v, off, 32));
            const float mnew = fmaxf(mrow[r], v);
            alpha[r] = exp2f(mrow[r] - mnew);
            mrow[r]  = mnew;
            lrow[r] *= alpha[r];
        }
#pragma unroll
        for (int nt = 0; nt < 4; ++nt)
#pragma unroll
            for (int r = 0; r < 8; ++r)
                oacc[nt][r] *= alpha[r];

        // ---- P = exp2(S - m) -> LDS (C-layout -> A-layout round trip) ----
#pragma unroll
        for (int nt = 0; nt < 4; ++nt)
#pragma unroll
            for (int r = 0; r < 8; ++r) {
                const float p = exp2f(sa[nt][r] - mrow[r]);
                lrow[r] += p;
                Ps[w][lh * 8 + r][nt * 16 + ll] = (_Float16)p;
            }

        // ---- O += P @ V ----
        v16h pf[2];
#pragma unroll
        for (int s = 0; s < 2; ++s)
#pragma unroll
            for (int e = 0; e < 16; ++e)
                pf[s][e] = Ps[w][ll][s * 32 + lh * 16 + e];
#pragma unroll
        for (int nt = 0; nt < 4; ++nt)
#pragma unroll
            for (int s = 0; s < 2; ++s) {
                v16h b;   // lane = d (N), element -> key (K); contiguous run
#pragma unroll
                for (int e = 0; e < 16; ++e)
                    b[e] = Vt[buf][nt * 16 + ll][s * 32 + lh * 16 + e];
                oacc[nt] = __builtin_amdgcn_wmma_f32_16x16x32_f16(
                    false, pf[s], false, b, (short)0, oacc[nt], false, false);
            }
        __syncthreads();
    }

    // ---- normalize and write o (f16, row-major [T, E]) ----
#pragma unroll
    for (int r = 0; r < 8; ++r) {
        float v = lrow[r];
#pragma unroll
        for (int off = 8; off >= 1; off >>= 1)
            v += __shfl_xor(v, off, 32);
        lrow[r] = 1.0f / v;
    }
#pragma unroll
    for (int nt = 0; nt < 4; ++nt)
#pragma unroll
        for (int r = 0; r < 8; ++r) {
            const int trow = qrow0 + lh * 8 + r;
            const int d    = nt * 16 + ll;
            oh[(size_t)trow * E_ + h * Dh + d] =
                (_Float16)(oacc[nt][r] * lrow[r]);
        }
}

// ------------------------------ launcher -----------------------------------
extern "C" void kernel_launch(void* const* d_in, const int* in_sizes, int n_in,
                              void* d_out, int out_size, void* d_ws, size_t ws_size,
                              hipStream_t stream)
{
    (void)in_sizes; (void)n_in; (void)out_size; (void)ws_size;
    const float* x     = (const float*)d_in[0];
    const float* freq  = (const float*)d_in[1];
    const float* w_qkv = (const float*)d_in[2];
    const float* w_out = (const float*)d_in[3];
    float* out = (float*)d_out;

    _Float16* ws    = (_Float16*)d_ws;
    _Float16* xh    = ws;                                // [T, E]
    _Float16* wTqkv = xh    + (size_t)T_ * E_;           // [QKVC, E]
    _Float16* wTout = wTqkv + (size_t)QKVC * E_;         // [E, E]
    _Float16* qkvh  = wTout + (size_t)E_ * E_;           // [T, QKVC]
    _Float16* vT    = qkvh  + (size_t)T_ * QKVC;         // [KDIM, T]
    _Float16* oh    = vT    + (size_t)KDIM_ * T_;        // [T, E]

    // one-time f16 conversion / weight transposes
    cvt_f16<<<(T_ * E_ / 4 + 255) / 256, 256, 0, stream>>>(x, xh, T_ * E_ / 4);
    transpose_cvt<<<dim3(QKVC / 32, E_ / 32), 256, 0, stream>>>(w_qkv, wTqkv, E_, QKVC);
    transpose_cvt<<<dim3(E_ / 32, E_ / 32), 256, 0, stream>>>(w_out, wTout, E_, E_);

    // 1) qkv = x @ w_qkv   (f16 out)
    wmma_gemm<QKVC, E_, true><<<dim3(QKVC / GTN, T_ / GTM), 256, 0, stream>>>(
        xh, wTqkv, qkvh);
    // 2) RoPE on q,k
    rope_f16<<<(T_ * ((E_ + KDIM_) / 2) + 255) / 256, 256, 0, stream>>>(qkvh, freq);
    // 3) v transpose for contiguous P@V fragments
    transpose_v<<<dim3(KDIM_ / 32, T_ / 32), 256, 0, stream>>>(qkvh, vT);
    // 4) causal GQA flash attention  (f16 out)
    flash_attn_wmma<<<dim3(T_ / QB, H_), 128, 0, stream>>>(qkvh, vT, oh);
    // 5) y = o @ w_out  (f32 out)
    wmma_gemm<E_, E_, false><<<dim3(E_ / GTN, T_ / GTM), 256, 0, stream>>>(
        oh, wTout, out);
}
